// GPT_11562051961543
// MI455X (gfx1250) — matrix operand order repaired and checked by reading, not compiled, not run on hardware
//
#include <hip/hip_runtime.h>
#include <math.h>

#define TT  1024
#define DD  768
#define NH  12
#define HSZ 64
#define NL  6
#define BB  2
#define FFD 3072
#define VV  32000

typedef __attribute__((ext_vector_type(16))) _Float16 v16h;
typedef __attribute__((ext_vector_type(8)))  _Float16 v8h;
typedef __attribute__((ext_vector_type(8)))  float    v8f;

__device__ __forceinline__ v8f wmma16(v16h a, v16h b, v8f c) {
  return __builtin_amdgcn_wmma_f32_16x16x32_f16(
      false, a, false, b, (short)0, c, false, false);
}

__device__ __forceinline__ v16h cat8(v8h lo, v8h hi) {
  return __builtin_shufflevector(lo, hi, 0, 1, 2, 3, 4, 5, 6, 7,
                                 8, 9, 10, 11, 12, 13, 14, 15);
}

// A-fragment (16x32 f16) from a row-major buffer (row stride ld halves).
// Lane layout (ISA 7.12.2): row = lane&15; K cols = g*8+(0..7) and 16+g*8+(0..7).
// Both chunks are 8 contiguous halves -> two aligned v8h loads.
__device__ __forceinline__ v16h load_afrag(const _Float16* base, int ld, int lane) {
  const int row = lane & 15, g = (lane >> 4) & 1;
  const _Float16* p = base + row * ld + g * 8;
  v8h lo = *(const v8h*)(p);
  v8h hi = *(const v8h*)(p + 16);
  return cat8(lo, hi);
}

// B-fragment (32x16 f16) from a K-major buffer: buf[n*ldn + k], k contiguous.
// Lane layout: n = lane&15; K = g*8+(0..7) and 16+g*8+(0..7) -> two aligned v8h loads.
__device__ __forceinline__ v16h load_bfrag(const _Float16* base, int ldn, int lane) {
  const int l15 = lane & 15, g = (lane >> 4) & 1;
  const _Float16* p = base + l15 * ldn + g * 8;
  return cat8(*(const v8h*)(p), *(const v8h*)(p + 16));
}

// ---------------------------------------------------------------------------
// Generic WMMA GEMM: C[M,N] = act(A[M,K] @ W[K,N] + bias)
// 64x64 C tile, 128 threads = 4 waves, each wave computes 16x64.
// Batched: z in grid.z, A += (z/nh)*aStride, W += (z%nh)*wStride, C += z*cStride.
// flags: bit0 = ReLU, bit1 = f16 out (Ch), bit2 = f16 out transposed (Ch[n*ldc+m]).
// All dims are multiples of the tile sizes for this model (no predication,
// EXEC stays all-ones as WMMA requires).
// ---------------------------------------------------------------------------
__global__ __launch_bounds__(128)
void gemm_kernel(const float* __restrict__ Abase, const float* __restrict__ Wbase,
                 const float* __restrict__ bias,
                 float* __restrict__ Cf, _Float16* __restrict__ Ch,
                 int M, int N, int K, int lda, int ldw, int ldc,
                 long aStride, long wStride, long cStride, int nh, int flags)
{
  __shared__ __align__(32) _Float16 sA[64 * 32];   // [m][k] row-major
  __shared__ __align__(32) _Float16 sBt[64 * 32];  // [n][k] K-major (transposed)

  const int z = blockIdx.z;
  const float* A = Abase + (long)(z / nh) * aStride;
  const float* W = Wbase + (long)(z % nh) * wStride;
  const long cbase = (long)z * cStride;

  const int m0 = blockIdx.y * 64;
  const int n0 = blockIdx.x * 64;
  const int tid = threadIdx.x;
  const int lane = tid & 31;
  const int wv = tid >> 5;
  const int half = (lane >> 4) & 1;
  const int l15 = lane & 15;

  // staging assignment: each thread covers one row (A) / one column (W),
  // half of the 32-wide K slab.
  const int r2 = tid >> 1;         // 0..63
  const int kb = (tid & 1) * 16;   // 0 or 16

  v8f acc[4] = {{}, {}, {}, {}};

  for (int k0 = 0; k0 < K; k0 += 32) {
    if (k0 + 32 < K) {  // -> global_prefetch_b8
      __builtin_prefetch(A + (long)(m0 + r2) * lda + k0 + 32 + kb, 0, 1);
      __builtin_prefetch(W + (long)(k0 + 32 + kb) * ldw + n0 + r2, 0, 1);
    }
    {
      // A tile: 16 contiguous f32 -> f16, two v8h LDS stores
      const float* ap = A + (long)(m0 + r2) * lda + k0 + kb;
      v8h a0, a1;
#pragma unroll
      for (int i = 0; i < 8; ++i) { a0[i] = (_Float16)ap[i]; a1[i] = (_Float16)ap[8 + i]; }
      *(v8h*)(sA + r2 * 32 + kb) = a0;
      *(v8h*)(sA + r2 * 32 + kb + 8) = a1;

      // W tile: 16 f32 strided by ldw (coalesced across lanes along n),
      // stored K-major (contiguous) in sBt
      const float* wp = W + (long)(k0 + kb) * ldw + n0 + r2;
      v8h b0, b1;
#pragma unroll
      for (int i = 0; i < 8; ++i) {
        b0[i] = (_Float16)wp[(long)i * ldw];
        b1[i] = (_Float16)wp[(long)(8 + i) * ldw];
      }
      *(v8h*)(sBt + r2 * 32 + kb) = b0;
      *(v8h*)(sBt + r2 * 32 + kb + 8) = b1;
    }
    __syncthreads();

    v16h af = load_afrag(sA + wv * 16 * 32, 32, lane);
#pragma unroll
    for (int nc = 0; nc < 4; ++nc) {
      v16h bf = load_bfrag(sBt + nc * 16 * 32, 32, lane);
      acc[nc] = wmma16(af, bf, acc[nc]);
    }
    __syncthreads();
  }

#pragma unroll
  for (int nc = 0; nc < 4; ++nc) {
#pragma unroll
    for (int r = 0; r < 8; ++r) {
      int m = m0 + wv * 16 + r + half * 8;
      int n = n0 + nc * 16 + l15;
      float val = acc[nc][r];
      if (bias) val += bias[n];
      if (flags & 1) val = fmaxf(val, 0.0f);
      if (flags & 4)      Ch[cbase + (long)n * ldc + m] = (_Float16)val;
      else if (flags & 2) Ch[cbase + (long)m * ldc + n] = (_Float16)val;
      else                Cf[cbase + (long)m * ldc + n] = val;
    }
  }
}

// ---------------------------------------------------------------------------
// Fused causal attention (flash-style).
// q,k: f16 [B,H,T,HS] row-major;  vT: f16 [B,H,HS,T] (transposed so PV
// B-fragments are contiguous);  o: f32 [B,T,D].
// 256 threads = 8 waves; each wave owns one 16-row Q tile; key slabs of 32
// so P (16x32) is exactly one WMMA A-operand. P staged per-wave through LDS
// with explicit s_wait_dscnt (single-wave producer/consumer).
// ---------------------------------------------------------------------------
__global__ __launch_bounds__(256)
void attn_kernel(const _Float16* __restrict__ q, const _Float16* __restrict__ k,
                 const _Float16* __restrict__ vT, float* __restrict__ o)
{
  __shared__ __align__(32) _Float16 sP[8 * 16 * 32];

  const int tid = threadIdx.x, lane = tid & 31, wv = tid >> 5;
  const int half = (lane >> 4) & 1;
  const int l15 = lane & 15;
  const int bh = blockIdx.y, b = bh / NH, h = bh % NH;
  const _Float16* qh = q  + (long)bh * TT * HSZ;
  const _Float16* kh = k  + (long)bh * TT * HSZ;
  const _Float16* vh = vT + (long)bh * HSZ * TT;
  const int m0 = (blockIdx.x * 8 + wv) * 16;

  // Q tile as two A-fragments (HS = 64 = 2 x K32), vector loads
  const v16h aQ0 = load_afrag(qh + (long)m0 * HSZ, HSZ, lane);
  const v16h aQ1 = load_afrag(qh + (long)m0 * HSZ + 32, HSZ, lane);

  v8f accO[4] = {{}, {}, {}, {}};
  float mi[8], li[8];
#pragma unroll
  for (int r = 0; r < 8; ++r) { mi[r] = -1e30f; li[r] = 0.f; }

  _Float16* myP = sP + wv * (16 * 32);
  const int jmax = (m0 + 15) >> 5;

  for (int j = 0; j <= jmax; ++j) {
    // S = Q * K^T over keys [32j, 32j+32): K rows are contiguous in the
    // needed (hs-contiguous) order -> two aligned v8h loads per fragment.
    v8f s0 = {}, s1 = {};
    {
      const _Float16* kb0 = kh + (long)(32 * j) * HSZ;       // keys 32j+l15
      const _Float16* kb1 = kh + (long)(32 * j + 16) * HSZ;  // keys 32j+16+l15
      s0 = wmma16(aQ0, load_bfrag(kb0, HSZ, lane), s0);
      s0 = wmma16(aQ1, load_bfrag(kb0 + 32, HSZ, lane), s0);
      s1 = wmma16(aQ0, load_bfrag(kb1, HSZ, lane), s1);
      s1 = wmma16(aQ1, load_bfrag(kb1 + 32, HSZ, lane), s1);
    }

    // online softmax over this 32-key slab (row stats via 16-lane shuffles)
    float al[8];
#pragma unroll
    for (int r = 0; r < 8; ++r) {
      const int m = m0 + r + half * 8;
      const int nk = 32 * j + l15;
      float v0 = (nk      <= m) ? s0[r] * 0.125f : -1e30f;  // HS^-0.5
      float v1 = (nk + 16 <= m) ? s1[r] * 0.125f : -1e30f;
      float t = fmaxf(v0, v1);
#pragma unroll
      for (int off = 1; off < 16; off <<= 1)
        t = fmaxf(t, __shfl_xor(t, off, 32));
      float newm = fmaxf(mi[r], t);
      float alpha = __expf(mi[r] - newm);
      float p0 = __expf(v0 - newm);
      float p1 = __expf(v1 - newm);
      float ps = p0 + p1;
#pragma unroll
      for (int off = 1; off < 16; off <<= 1)
        ps += __shfl_xor(ps, off, 32);
      li[r] = li[r] * alpha + ps;
      mi[r] = newm;
      al[r] = alpha;
      myP[(r + half * 8) * 32 + l15] = (_Float16)p0;
      myP[(r + half * 8) * 32 + 16 + l15] = (_Float16)p1;
    }
#pragma unroll
    for (int nc = 0; nc < 4; ++nc)
#pragma unroll
      for (int r = 0; r < 8; ++r)
        accO[nc][r] *= al[r];

    // single-wave LDS producer->consumer: DS counter only, no block barrier
    asm volatile("s_wait_dscnt 0" ::: "memory");

    v16h aP = load_afrag(myP, 32, lane);

    // O += P * V : V^T layout makes each B-fragment k-contiguous (two v8h loads per lane)
#pragma unroll
    for (int nc = 0; nc < 4; ++nc) {
      v16h bv = load_bfrag(vh + (long)(nc * 16) * TT + 32 * j, TT, lane);
      accO[nc] = wmma16(aP, bv, accO[nc]);
    }
    asm volatile("s_wait_dscnt 0" ::: "memory");  // reads done before overwrite
  }

#pragma unroll
  for (int r = 0; r < 8; ++r) li[r] = 1.0f / li[r];
#pragma unroll
  for (int nc = 0; nc < 4; ++nc)
#pragma unroll
    for (int r = 0; r < 8; ++r) {
      int m = m0 + r + half * 8;
      int c = h * HSZ + nc * 16 + l15;
      o[((long)b * TT + m) * DD + c] = accO[nc][r] * li[r];
    }
}

// ---------------------------------------------------------------------------
// out[row,:] = (resid ? resid[row,:] : 0) + LayerNorm(a[row,:]) * g + b
// ---------------------------------------------------------------------------
__global__ __launch_bounds__(256)
void lnadd_kernel(const float* __restrict__ resid, const float* __restrict__ a,
                  const float* __restrict__ g, const float* __restrict__ bt,
                  float* __restrict__ out)
{
  __shared__ float rs_[256], rq_[256];
  const int row = blockIdx.x, tid = threadIdx.x;
  const float* ar = a + (long)row * DD;
  float s = 0.f, qq = 0.f, vals[3];
#pragma unroll
  for (int i = 0; i < 3; ++i) {
    float t = ar[tid + i * 256];
    vals[i] = t; s += t; qq += t * t;
  }
  rs_[tid] = s; rq_[tid] = qq;
  __syncthreads();
  for (int st = 128; st > 0; st >>= 1) {
    if (tid < st) { rs_[tid] += rs_[tid + st]; rq_[tid] += rq_[tid + st]; }
    __syncthreads();
  }
  float mean = rs_[0] * (1.0f / DD);
  float var  = rq_[0] * (1.0f / DD) - mean * mean;
  float rstd = rsqrtf(var + 1e-5f);
#pragma unroll
  for (int i = 0; i < 3; ++i) {
    int c = tid + i * 256;
    float val = (vals[i] - mean) * rstd * g[c] + bt[c];
    if (resid) val += resid[(long)row * DD + c];
    out[(long)row * DD + c] = val;
  }
}

__global__ __launch_bounds__(256)
void embed_kernel(const int* __restrict__ idx, const float* __restrict__ tok,
                  const float* __restrict__ pos, float* __restrict__ x)
{
  int bt = blockIdx.x;
  int t = bt % TT;
  int token = idx[bt];
  for (int i = threadIdx.x; i < DD; i += 256)
    x[(long)bt * DD + i] = tok[(long)token * DD + i] + pos[(long)t * DD + i];
}

// ---------------------------------------------------------------------------
extern "C" void kernel_launch(void* const* d_in, const int* in_sizes, int n_in,
                              void* d_out, int out_size, void* d_ws, size_t ws_size,
                              hipStream_t stream)
{
  (void)in_sizes; (void)n_in; (void)out_size; (void)ws_size;
  const int*   idx = (const int*)d_in[0];
  const float* tok = (const float*)d_in[1];
  const float* pos = (const float*)d_in[2];
  const float* Wq  = (const float*)d_in[3];
  const float* Wk  = (const float*)d_in[4];
  const float* Wv  = (const float*)d_in[5];
  const float* Wo  = (const float*)d_in[6];
  const float* bo  = (const float*)d_in[7];
  const float* W1  = (const float*)d_in[8];
  const float* b1  = (const float*)d_in[9];
  const float* W2  = (const float*)d_in[10];
  const float* b2  = (const float*)d_in[11];
  const float* g1  = (const float*)d_in[12];
  const float* be1 = (const float*)d_in[13];
  const float* g2  = (const float*)d_in[14];
  const float* be2 = (const float*)d_in[15];
  const float* gf  = (const float*)d_in[16];
  const float* bef = (const float*)d_in[17];
  const float* Wlm = (const float*)d_in[18];
  const float* blm = (const float*)d_in[19];
  float* out = (float*)d_out;

  const size_t szBTD  = (size_t)BB * TT * DD * sizeof(float);
  const size_t szBTDh = (size_t)BB * TT * DD * sizeof(_Float16);
  const size_t szFF   = (size_t)BB * TT * FFD * sizeof(float);
  char* ws = (char*)d_ws;
  float*    x    = (float*)ws;    ws += szBTD;
  float*    abuf = (float*)ws;    ws += szBTD;
  float*    hbuf = (float*)ws;    ws += szBTD;
  float*    obuf = (float*)ws;    ws += szBTD;
  float*    ff   = (float*)ws;    ws += szFF;
  _Float16* qb   = (_Float16*)ws; ws += szBTDh;
  _Float16* kb   = (_Float16*)ws; ws += szBTDh;
  _Float16* vb   = (_Float16*)ws; ws += szBTDh;   // stored transposed [B,H,HS,T]

  const int MBT = BB * TT;

  embed_kernel<<<BB * TT, 256, 0, stream>>>(idx, tok, pos, x);

  for (int l = 0; l < NL; ++l) {
    const long wL = (long)l * NH * DD * HSZ;
    dim3 gq(1, TT / 64, BB * NH);
    // Q, K: f16 [B,H,T,HS]
    gemm_kernel<<<gq, 128, 0, stream>>>(x, Wq + wL, nullptr, nullptr, qb,
        TT, HSZ, DD, DD, HSZ, HSZ,
        (long)TT * DD, (long)DD * HSZ, (long)TT * HSZ, NH, 2);
    gemm_kernel<<<gq, 128, 0, stream>>>(x, Wk + wL, nullptr, nullptr, kb,
        TT, HSZ, DD, DD, HSZ, HSZ,
        (long)TT * DD, (long)DD * HSZ, (long)TT * HSZ, NH, 2);
    // V: f16 transposed [B,H,HS,T]  (flags 2|4, ldc = T)
    gemm_kernel<<<gq, 128, 0, stream>>>(x, Wv + wL, nullptr, nullptr, vb,
        TT, HSZ, DD, DD, HSZ, TT,
        (long)TT * DD, (long)DD * HSZ, (long)HSZ * TT, NH, 6);

    attn_kernel<<<dim3(TT / 128, BB * NH), 256, 0, stream>>>(qb, kb, vb, obuf);

    gemm_kernel<<<dim3(DD / 64, MBT / 64, 1), 128, 0, stream>>>(
        obuf, Wo + (long)l * DD * DD, bo + (long)l * DD, abuf, nullptr,
        MBT, DD, DD, DD, DD, DD, 0, 0, 0, 1, 0);

    lnadd_kernel<<<MBT, 256, 0, stream>>>(x, abuf, g1 + (long)l * DD,
                                          be1 + (long)l * DD, hbuf);

    gemm_kernel<<<dim3(FFD / 64, MBT / 64, 1), 128, 0, stream>>>(
        hbuf, W1 + (long)l * DD * FFD, b1 + (long)l * FFD, ff, nullptr,
        MBT, FFD, DD, DD, FFD, FFD, 0, 0, 0, 1, 1);

    gemm_kernel<<<dim3(DD / 64, MBT / 64, 1), 128, 0, stream>>>(
        ff, W2 + (long)l * FFD * DD, b2 + (long)l * DD, abuf, nullptr,
        MBT, DD, FFD, FFD, DD, DD, 0, 0, 0, 1, 0);

    // x = x + ln2(ff_out)  (residual from ORIGINAL x, matching reference)
    lnadd_kernel<<<MBT, 256, 0, stream>>>(x, abuf, g2 + (long)l * DD,
                                          be2 + (long)l * DD, x);
  }

  lnadd_kernel<<<MBT, 256, 0, stream>>>(nullptr, x, gf, bef, hbuf);
  gemm_kernel<<<dim3(VV / 64, MBT / 64, 1), 128, 0, stream>>>(
      hbuf, Wlm, blm, out, nullptr,
      MBT, VV, DD, DD, VV, VV, 0, 0, 0, 1, 0);
}
